// TemporalConvVQVAE_15710990369636
// MI455X (gfx1250) — compile-verified, matrix operand order checked by reading, not run on hardware
//
#include <hip/hip_runtime.h>
#include <hip/hip_bf16.h>
#include <math.h>

typedef __attribute__((ext_vector_type(16))) __bf16 v16bf;
typedef __attribute__((ext_vector_type(8)))  float  v8f;

union FragU { uint4 q[2]; v16bf v; };

__device__ inline unsigned short f2bf(float f) {
    unsigned int u = __float_as_uint(f);
    unsigned int r = (u + 0x7FFFu + ((u >> 16) & 1u)) >> 16;
    return (unsigned short)r;
}

// Load a 16-bf16 WMMA fragment: two contiguous 16B chunks at k and k+16
// (caller has already applied the per-lane ksel offset to p).
__device__ inline v16bf frag2(const unsigned short* p) {
    FragU f;
    f.q[0] = *(const uint4*)(p);
    f.q[1] = *(const uint4*)(p + 16);
    return f.v;
}

__device__ inline v8f wmma_bf16(v16bf a, v16bf b, v8f c) {
    return __builtin_amdgcn_wmma_f32_16x16x32_bf16(
        /*neg_a=*/false, a, /*neg_b=*/false, b,
        /*c_mod=*/(short)0, c, /*reuse_a=*/false, /*reuse_b=*/false);
}

// ---------------- constants ----------------
#define BTOT   65536
#define TT     10
#define CIN    142
#define TILEB  16

// output layout (floats)
#define VP_OFF  0L
#define ZE_OFF  131072L
#define ZQ_OFF  8519680L
#define IDX_OFF 16908288L
#define CB_OFF  16973824L
#define CM_OFF  16973825L
#define PP_OFF  16973826L

// workspace layout (bytes)
#define WS_WT1   0          // [256][480] bf16
#define WS_WT2   245760     // [128][768] bf16
#define WS_FCW   442368     // [128][128] bf16
#define WS_CBW   475136     // [512][128] bf16
#define WS_D1W   606208     // [256][128] bf16
#define WS_D2W   671744     // [128][256] bf16
#define WS_CN    737280     // [512] f32
#define WS_SC1   739328
#define WS_SH1   740352
#define WS_SC2   741376
#define WS_SH2   741888
#define WS_HIST  742400     // [512] u32
#define WS_GSUM  744448     // f32

// LDS layout (bytes), dynamic
#define SM_XT    0          // [16][12][160] bf16  (zero-padded t and i)
#define SM_H1T   61440      // [16][12][256] bf16
#define SM_FBUF  159744     // [16][512] f32  (reused scratch)
#define SM_ABA   192512     // [16][256] bf16 (A-matrix staging, ping)
#define SM_ABB   200704     // [16][256] bf16 (A-matrix staging, pong)
#define SM_MU    208896
#define SM_RS    208960
#define SM_ZN    209024
#define SM_IDX   209088
#define SM_BSUM  209152
#define SM_BYTES 209408

// ---------------- prep kernels ----------------
__global__ void k_zero(unsigned int* hist, float* gsum) {
    int i = threadIdx.x;
    if (i < 512) hist[i] = 0u;
    if (i == 0) *gsum = 0.f;
}

__global__ void k_pack_conv1(const float* __restrict__ w, unsigned short* __restrict__ o16) {
    int idx = blockIdx.x * 256 + threadIdx.x;
    if (idx >= 256 * 480) return;
    int o = idx / 480, k = idx - o * 480;
    int kk = k / 160, i = k - kk * 160;
    float v = (i < CIN) ? w[(o * CIN + i) * 3 + kk] : 0.f;
    o16[idx] = f2bf(v);
}

__global__ void k_pack_conv2(const float* __restrict__ w, unsigned short* __restrict__ o16) {
    int idx = blockIdx.x * 256 + threadIdx.x;
    if (idx >= 128 * 768) return;
    int o = idx / 768, k = idx - o * 768;
    int kk = k >> 8, i = k & 255;
    o16[idx] = f2bf(w[(o * 256 + i) * 3 + kk]);
}

__global__ void k_pack_mat(const float* __restrict__ w, unsigned short* __restrict__ o16, int n) {
    int idx = blockIdx.x * 256 + threadIdx.x;
    if (idx < n) o16[idx] = f2bf(w[idx]);
}

__global__ void k_cbnorm(const float* __restrict__ cb, float* __restrict__ cn) {
    int c = blockIdx.x * 256 + threadIdx.x;
    if (c < 512) {
        float s = 0.f;
        for (int j = 0; j < 128; ++j) { float v = cb[c * 128 + j]; s += v * v; }
        cn[c] = s;
    }
}

__global__ void k_prep_bn(const float* g, const float* b, const float* m, const float* v,
                          const float* bias, float* sc, float* sh, int n) {
    int o = blockIdx.x * 256 + threadIdx.x;
    if (o < n) {
        float s = g[o] * rsqrtf(v[o] + 1e-5f);
        sc[o] = s;
        sh[o] = bias[o] * s + b[o] - m[o] * s;
    }
}

// ---------------- epilogue helpers ----------------
__device__ inline void conv1_epi(v8f acc, int o, int t, int mhal,
                                 const float* sc1, const float* sh1, unsigned short* h1t) {
    float sc = sc1[o], sh = sh1[o];
#pragma unroll
    for (int r = 0; r < 8; ++r) {
        float f = acc[r] * sc + sh;
        f = f > 0.f ? f : 0.f;
        h1t[((r + mhal) * 12 + t + 1) * 256 + o] = f2bf(f);
    }
}

// ---------------- fused mega kernel ----------------
__global__ __launch_bounds__(128) void vqvae_mega(
    const float* __restrict__ x,
    const unsigned short* __restrict__ Wt1, const float* __restrict__ sc1, const float* __restrict__ sh1,
    const unsigned short* __restrict__ Wt2, const float* __restrict__ sc2, const float* __restrict__ sh2,
    const unsigned short* __restrict__ fcW, const float* __restrict__ fc_b,
    const float* __restrict__ ln1_g, const float* __restrict__ ln1_b,
    const unsigned short* __restrict__ cbW, const float* __restrict__ cnorm,
    const float* __restrict__ codebook,
    const unsigned short* __restrict__ d1W, const float* __restrict__ dec1_b,
    const float* __restrict__ ln2_g, const float* __restrict__ ln2_b,
    const unsigned short* __restrict__ d2W, const float* __restrict__ dec2_b,
    const float* __restrict__ dec3_w, const float* __restrict__ dec3_b,
    unsigned int* __restrict__ hist, float* __restrict__ gsum,
    float* __restrict__ out)
{
    extern __shared__ char smem[];
    unsigned short* xt  = (unsigned short*)(smem + SM_XT);
    unsigned short* h1t = (unsigned short*)(smem + SM_H1T);
    float*          fbuf = (float*)(smem + SM_FBUF);
    unsigned short* abA = (unsigned short*)(smem + SM_ABA);
    unsigned short* abB = (unsigned short*)(smem + SM_ABB);
    float* muv  = (float*)(smem + SM_MU);
    float* rsv  = (float*)(smem + SM_RS);
    float* znv  = (float*)(smem + SM_ZN);
    int*   idxs = (int*)(smem + SM_IDX);
    float* bsum = (float*)(smem + SM_BSUM);
    float* zef  = (float*)(smem + SM_XT);   // reuse xt region after conv1 for z_e f32 [16][128]

    const int tid  = threadIdx.x;
    const int lane = tid & 31;
    const int wave = tid >> 5;
    const int b0   = blockIdx.x * TILEB;
    const int mrow = lane & 15;            // M (or N) index of this lane within a tile
    const int ksel = (lane >> 4) << 3;     // 0 / 8 : per-lane K sub-offset (ISA 16-bit A/B layout)
    const int mhal = (lane >> 4) << 3;     // 0 / 8 : C/D row offset

    // ---- phase 0: zero xt & h1t, init bsum, fill xt (bf16, t-padded) ----
    for (int e = tid; e < (61440 + 98304) / 4; e += 128) ((unsigned int*)smem)[e] = 0u;
    if (tid == 0) *bsum = 0.f;
    __syncthreads();
    for (int e = tid; e < TILEB * TT * CIN; e += 128) {
        int b = e / (TT * CIN);
        int rem = e - b * (TT * CIN);
        int t = rem / CIN, i = rem - t * CIN;
        float v = x[((long)(b0 + b) * TT + t) * CIN + i];
        xt[(b * 12 + t + 1) * 160 + i] = f2bf(v);
    }
    __syncthreads();

    // ---- phase 1: conv1 + BN + ReLU -> h1t ----
    // M = 16 batches at fixed t; N = 256 (4 N-tiles per wave); K = 480 (15 steps)
    for (int t = 0; t < TT; ++t) {
        v8f a0{}, a1{}, a2{}, a3{};
        for (int ks = 0; ks < 15; ++ks) {
            int kk = ks / 5;
            int i0 = ks * 32 + ksel - kk * 160;
            v16bf af = frag2(&xt[(mrow * 12 + t + kk) * 160 + i0]);
            const unsigned short* bp = &Wt1[((wave * 4) * 16 + mrow) * 480 + ks * 32 + ksel];
            a0 = wmma_bf16(af, frag2(bp),            a0);
            a1 = wmma_bf16(af, frag2(bp + 16 * 480), a1);
            a2 = wmma_bf16(af, frag2(bp + 32 * 480), a2);
            a3 = wmma_bf16(af, frag2(bp + 48 * 480), a3);
        }
        int ob = (wave * 4) * 16 + mrow;
        conv1_epi(a0, ob,      t, mhal, sc1, sh1, h1t);
        conv1_epi(a1, ob + 16, t, mhal, sc1, sh1, h1t);
        conv1_epi(a2, ob + 32, t, mhal, sc1, sh1, h1t);
        conv1_epi(a3, ob + 48, t, mhal, sc1, sh1, h1t);
    }
    __syncthreads();

    // ---- phase 2: conv2 + BN + ReLU + mean over T ----
    // N = 128 (2 N-tiles per wave); K = 768 (24 steps)
    v8f pool0{}, pool1{};
    for (int t = 0; t < TT; ++t) {
        v8f c0{}, c1{};
        for (int ks = 0; ks < 24; ++ks) {
            int kk = ks >> 3;
            int i0 = ((ks & 7) << 5) + ksel;
            v16bf af = frag2(&h1t[(mrow * 12 + t + kk) * 256 + i0]);
            const unsigned short* bp = &Wt2[((wave * 2) * 16 + mrow) * 768 + ks * 32 + ksel];
            c0 = wmma_bf16(af, frag2(bp),            c0);
            c1 = wmma_bf16(af, frag2(bp + 16 * 768), c1);
        }
        int o0 = (wave * 2) * 16 + mrow;
        float s0 = sc2[o0],      h0 = sh2[o0];
        float s1 = sc2[o0 + 16], h1v = sh2[o0 + 16];
#pragma unroll
        for (int r = 0; r < 8; ++r) {
            float f0 = c0[r] * s0 + h0;  pool0[r] += (f0 > 0.f ? f0 : 0.f);
            float f1 = c1[r] * s1 + h1v; pool1[r] += (f1 > 0.f ? f1 : 0.f);
        }
    }
#pragma unroll
    for (int r = 0; r < 8; ++r) {
        int m = r + mhal;
        abA[m * 128 + (wave * 2) * 16 + mrow]       = f2bf(pool0[r] * 0.1f);
        abA[m * 128 + (wave * 2 + 1) * 16 + mrow]   = f2bf(pool1[r] * 0.1f);
    }
    __syncthreads();

    // ---- phase 3: fc GEMM (M16,N128,K128) + LayerNorm + ReLU -> z_e ----
    {
        v8f f0{}, f1{};
        for (int ks = 0; ks < 4; ++ks) {
            v16bf af = frag2(&abA[mrow * 128 + ks * 32 + ksel]);
            const unsigned short* bp = &fcW[((wave * 2) * 16 + mrow) * 128 + ks * 32 + ksel];
            f0 = wmma_bf16(af, frag2(bp),            f0);
            f1 = wmma_bf16(af, frag2(bp + 16 * 128), f1);
        }
        int n0 = (wave * 2) * 16 + mrow;
        float bb0 = fc_b[n0], bb1 = fc_b[n0 + 16];
#pragma unroll
        for (int r = 0; r < 8; ++r) {
            int m = r + mhal;
            fbuf[m * 128 + n0]      = f0[r] + bb0;
            fbuf[m * 128 + n0 + 16] = f1[r] + bb1;
        }
    }
    __syncthreads();
    if (tid < 16) {
        float s = 0.f, s2 = 0.f;
        for (int j = 0; j < 128; ++j) { float v = fbuf[tid * 128 + j]; s += v; s2 += v * v; }
        float mu = s * (1.f / 128.f);
        float var = s2 * (1.f / 128.f) - mu * mu;
        muv[tid] = mu; rsv[tid] = rsqrtf(var + 1e-5f);
    }
    __syncthreads();
    for (int e = tid; e < 2048; e += 128) {
        int r = e >> 7, j = e & 127;
        float v = (fbuf[e] - muv[r]) * rsv[r] * ln1_g[j] + ln1_b[j];
        v = v > 0.f ? v : 0.f;
        zef[e] = v;
        abB[r * 128 + j] = f2bf(v);
        out[ZE_OFF + (long)(b0 + r) * 128 + j] = v;
    }
    __syncthreads();
    if (tid < 16) {
        float s = 0.f;
        for (int j = 0; j < 128; ++j) { float v = zef[tid * 128 + j]; s += v * v; }
        znv[tid] = s;
    }
    __syncthreads();

    // ---- phase 4: VQ distance GEMM (M16,N512,K128), argmin, z_q_st, losses ----
    for (int q = 0; q < 8; ++q) {
        int n0 = (wave * 8 + q) * 16 + mrow;     // codebook row
        v8f dacc{};
        for (int ks = 0; ks < 4; ++ks) {
            v16bf af = frag2(&abB[mrow * 128 + ks * 32 + ksel]);
            dacc = wmma_bf16(af, frag2(&cbW[n0 * 128 + ks * 32 + ksel]), dacc);
        }
        float cn = cnorm[n0];
#pragma unroll
        for (int r = 0; r < 8; ++r) {
            int m = r + mhal;
            fbuf[m * 512 + n0] = znv[m] - 2.f * dacc[r] + cn;
        }
    }
    __syncthreads();
    if (tid < 16) {
        float bv = 3.4e38f; int bi = 0;
        for (int c = 0; c < 512; ++c) {
            float v = fbuf[tid * 512 + c];
            if (v < bv) { bv = v; bi = c; }
        }
        idxs[tid] = bi;
        out[IDX_OFF + b0 + tid] = (float)bi;
        atomicAdd(&hist[bi], 1u);
    }
    __syncthreads();
    {
        float ls = 0.f;
        for (int e = tid; e < 2048; e += 128) {
            int r = e >> 7, j = e & 127;
            float ze = zef[e];
            float zq = codebook[idxs[r] * 128 + j];
            float d = zq - ze; ls += d * d;
            float zst = ze + (zq - ze);
            out[ZQ_OFF + (long)(b0 + r) * 128 + j] = zst;
            abA[e] = f2bf(zst);
        }
        atomicAdd(bsum, ls);
    }
    __syncthreads();
    if (tid == 0) atomicAdd(gsum, *bsum);

    // ---- phase 5: dec1 GEMM (M16,N256,K128) + LN + ReLU ----
    for (int q = 0; q < 4; ++q) {
        int n0 = (wave * 4 + q) * 16 + mrow;
        v8f acc{};
        for (int ks = 0; ks < 4; ++ks) {
            v16bf af = frag2(&abA[mrow * 128 + ks * 32 + ksel]);
            acc = wmma_bf16(af, frag2(&d1W[n0 * 128 + ks * 32 + ksel]), acc);
        }
        float bb = dec1_b[n0];
#pragma unroll
        for (int r = 0; r < 8; ++r) fbuf[(r + mhal) * 256 + n0] = acc[r] + bb;
    }
    __syncthreads();
    if (tid < 16) {
        float s = 0.f, s2 = 0.f;
        for (int j = 0; j < 256; ++j) { float v = fbuf[tid * 256 + j]; s += v; s2 += v * v; }
        float mu = s * (1.f / 256.f);
        float var = s2 * (1.f / 256.f) - mu * mu;
        muv[tid] = mu; rsv[tid] = rsqrtf(var + 1e-5f);
    }
    __syncthreads();
    for (int e = tid; e < 4096; e += 128) {
        int r = e >> 8, j = e & 255;
        float v = (fbuf[e] - muv[r]) * rsv[r] * ln2_g[j] + ln2_b[j];
        v = v > 0.f ? v : 0.f;
        abB[e] = f2bf(v);
    }
    __syncthreads();

    // ---- phase 6: dec2 GEMM (M16,N128,K256) + ReLU ----
    {
        v8f g0{}, g1{};
        for (int ks = 0; ks < 8; ++ks) {
            v16bf af = frag2(&abB[mrow * 256 + ks * 32 + ksel]);
            const unsigned short* bp = &d2W[((wave * 2) * 16 + mrow) * 256 + ks * 32 + ksel];
            g0 = wmma_bf16(af, frag2(bp),            g0);
            g1 = wmma_bf16(af, frag2(bp + 16 * 256), g1);
        }
        int n0 = (wave * 2) * 16 + mrow;
        float b00 = dec2_b[n0], b11 = dec2_b[n0 + 16];
#pragma unroll
        for (int r = 0; r < 8; ++r) {
            int m = r + mhal;
            float v0 = g0[r] + b00; fbuf[m * 128 + n0]      = v0 > 0.f ? v0 : 0.f;
            float v1 = g1[r] + b11; fbuf[m * 128 + n0 + 16] = v1 > 0.f ? v1 : 0.f;
        }
    }
    __syncthreads();

    // ---- phase 7: dec3 (128 -> 2) ----
    if (tid < 32) {
        int r = tid >> 1, c = tid & 1;
        float s = dec3_b[c];
        for (int k = 0; k < 128; ++k) s += fbuf[r * 128 + k] * dec3_w[c * 128 + k];
        out[VP_OFF + (long)(b0 + r) * 2 + c] = s;
    }
}

// ---------------- finalize: scalar losses + perplexity ----------------
__global__ __launch_bounds__(512) void k_final(const unsigned int* hist, const float* gsum, float* out) {
    __shared__ float red[512];
    int c = threadIdx.x;
    float p = (float)hist[c] * (1.f / 65536.f);
    red[c] = p * logf(p + 1e-10f);
    __syncthreads();
    for (int s = 256; s > 0; s >>= 1) {
        if (c < s) red[c] += red[c + s];
        __syncthreads();
    }
    if (c == 0) {
        float lm = (*gsum) * (1.f / (65536.f * 128.f));
        out[CB_OFF] = lm;
        out[CM_OFF] = 0.1f * lm;
        out[PP_OFF] = expf(-red[0]);
    }
}

extern "C" void kernel_launch(void* const* d_in, const int* in_sizes, int n_in,
                              void* d_out, int out_size, void* d_ws, size_t ws_size,
                              hipStream_t stream) {
    (void)in_sizes; (void)n_in; (void)out_size; (void)ws_size;
    const float* x        = (const float*)d_in[0];
    const float* conv1_w  = (const float*)d_in[1];
    const float* conv1_b  = (const float*)d_in[2];
    const float* bn1_g    = (const float*)d_in[3];
    const float* bn1_b    = (const float*)d_in[4];
    const float* bn1_m    = (const float*)d_in[5];
    const float* bn1_v    = (const float*)d_in[6];
    const float* conv2_w  = (const float*)d_in[7];
    const float* conv2_b  = (const float*)d_in[8];
    const float* bn2_g    = (const float*)d_in[9];
    const float* bn2_b    = (const float*)d_in[10];
    const float* bn2_m    = (const float*)d_in[11];
    const float* bn2_v    = (const float*)d_in[12];
    const float* fc_w     = (const float*)d_in[13];
    const float* fc_b     = (const float*)d_in[14];
    const float* ln1_g    = (const float*)d_in[15];
    const float* ln1_b    = (const float*)d_in[16];
    const float* codebook = (const float*)d_in[17];
    const float* dec1_w   = (const float*)d_in[18];
    const float* dec1_b   = (const float*)d_in[19];
    const float* ln2_g    = (const float*)d_in[20];
    const float* ln2_b    = (const float*)d_in[21];
    const float* dec2_w   = (const float*)d_in[22];
    const float* dec2_b   = (const float*)d_in[23];
    const float* dec3_w   = (const float*)d_in[24];
    const float* dec3_b   = (const float*)d_in[25];

    char* ws = (char*)d_ws;
    unsigned short* Wt1 = (unsigned short*)(ws + WS_WT1);
    unsigned short* Wt2 = (unsigned short*)(ws + WS_WT2);
    unsigned short* fcW = (unsigned short*)(ws + WS_FCW);
    unsigned short* cbW = (unsigned short*)(ws + WS_CBW);
    unsigned short* d1W = (unsigned short*)(ws + WS_D1W);
    unsigned short* d2W = (unsigned short*)(ws + WS_D2W);
    float* cnorm = (float*)(ws + WS_CN);
    float* sc1 = (float*)(ws + WS_SC1);
    float* sh1 = (float*)(ws + WS_SH1);
    float* sc2 = (float*)(ws + WS_SC2);
    float* sh2 = (float*)(ws + WS_SH2);
    unsigned int* hist = (unsigned int*)(ws + WS_HIST);
    float* gsum = (float*)(ws + WS_GSUM);
    float* out  = (float*)d_out;

    k_zero<<<1, 512, 0, stream>>>(hist, gsum);
    k_pack_conv1<<<(256 * 480 + 255) / 256, 256, 0, stream>>>(conv1_w, Wt1);
    k_pack_conv2<<<(128 * 768 + 255) / 256, 256, 0, stream>>>(conv2_w, Wt2);
    k_pack_mat<<<(128 * 128 + 255) / 256, 256, 0, stream>>>(fc_w, fcW, 128 * 128);
    k_pack_mat<<<(512 * 128 + 255) / 256, 256, 0, stream>>>(codebook, cbW, 512 * 128);
    k_pack_mat<<<(256 * 128 + 255) / 256, 256, 0, stream>>>(dec1_w, d1W, 256 * 128);
    k_pack_mat<<<(128 * 256 + 255) / 256, 256, 0, stream>>>(dec2_w, d2W, 128 * 256);
    k_cbnorm<<<2, 256, 0, stream>>>(codebook, cnorm);
    k_prep_bn<<<1, 256, 0, stream>>>(bn1_g, bn1_b, bn1_m, bn1_v, conv1_b, sc1, sh1, 256);
    k_prep_bn<<<1, 256, 0, stream>>>(bn2_g, bn2_b, bn2_m, bn2_v, conv2_b, sc2, sh2, 128);

    vqvae_mega<<<BTOT / TILEB, 128, SM_BYTES, stream>>>(
        x, Wt1, sc1, sh1, Wt2, sc2, sh2, fcW, fc_b, ln1_g, ln1_b,
        cbW, cnorm, codebook, d1W, dec1_b, ln2_g, ln2_b,
        d2W, dec2_b, dec3_w, dec3_b, hist, gsum, out);

    k_final<<<1, 512, 0, stream>>>(hist, gsum, out);
}
